// MultiHeadAttention_76484777607720
// MI455X (gfx1250) — matrix operand order repaired and checked by reading, not compiled, not run on hardware
//
#include <hip/hip_runtime.h>
#include <hip/hip_bf16.h>

#define D_MODEL 1024
#define N_HEAD  16
#define D_K     64
#define BATCH   4
#define SEQ     2048
#define BS      (BATCH * SEQ)   // 8192 rows

typedef __attribute__((ext_vector_type(16))) _Float16 v16h;
typedef __attribute__((ext_vector_type(8)))  _Float16 v8h;
typedef __attribute__((ext_vector_type(4)))  _Float16 v4h;
typedef __attribute__((ext_vector_type(8)))  float    v8f;

static __device__ __forceinline__ v8f wmma_f16(v16h a, v16h b, v8f c) {
    // D = A(16x32 f16) * B(32x16 f16) + C(16x16 f32)
    return __builtin_amdgcn_wmma_f32_16x16x32_f16(
        /*neg_a=*/false, a, /*neg_b=*/false, b,
        /*c_mod=*/(short)0, c, /*reuse_a=*/false, /*reuse_b=*/false);
}

// A fragment 16x32 (f16 source): lane l -> row = l&15, K chunks {kb..kb+7, kb+16..kb+23}, kb = 8*(l>>4)
static __device__ __forceinline__ v16h frag_a_f16(const _Float16* p, int ld, int l) {
    int row = l & 15, kb = (l >> 4) * 8;
    const _Float16* q = p + (size_t)row * ld;
    v8h lo = *(const v8h*)(q + kb);
    v8h hi = *(const v8h*)(q + kb + 16);
    v16h r;
#pragma unroll
    for (int i = 0; i < 8; ++i) { r[i] = lo[i]; r[i + 8] = hi[i]; }
    return r;
}

// A fragment 16x32 sourced from fp32 (converted in-register)
static __device__ __forceinline__ v16h frag_a_f32(const float* p, int ld, int l) {
    int row = l & 15, kb = (l >> 4) * 8;
    const float* q = p + (size_t)row * ld;
    float4 a0 = *(const float4*)(q + kb);
    float4 a1 = *(const float4*)(q + kb + 4);
    float4 b0 = *(const float4*)(q + kb + 16);
    float4 b1 = *(const float4*)(q + kb + 20);
    v16h r;
    r[0]=(_Float16)a0.x; r[1]=(_Float16)a0.y; r[2]=(_Float16)a0.z; r[3]=(_Float16)a0.w;
    r[4]=(_Float16)a1.x; r[5]=(_Float16)a1.y; r[6]=(_Float16)a1.z; r[7]=(_Float16)a1.w;
    r[8]=(_Float16)b0.x; r[9]=(_Float16)b0.y; r[10]=(_Float16)b0.z; r[11]=(_Float16)b0.w;
    r[12]=(_Float16)b1.x; r[13]=(_Float16)b1.y; r[14]=(_Float16)b1.z; r[15]=(_Float16)b1.w;
    return r;
}

// B fragment 32x16: lane l -> col n = l&15, K chunks {kb..kb+7, kb+16..kb+23}, kb = 8*(l>>4).
// Storage convention: element (k, n) lives at p[n*ld + k]  (i.e. B^T row-major)
static __device__ __forceinline__ v16h frag_b_f16(const _Float16* p, int ld, int l) {
    int col = l & 15, kb = (l >> 4) * 8;
    const _Float16* q = p + (size_t)col * ld + kb;
    v8h lo = *(const v8h*)(q);
    v8h hi = *(const v8h*)(q + 16);
    v16h r;
#pragma unroll
    for (int i = 0; i < 8; ++i) { r[i] = lo[i]; r[i + 8] = hi[i]; }
    return r;
}

// ---------------------------------------------------------------- kernel 1
// fp32 -> f16 weight conversion (1M elements per matrix), vectorized x4
__global__ __launch_bounds__(256) void convert_w_kernel(const float* __restrict__ src,
                                                        _Float16* __restrict__ dst, int n4) {
    int i = blockIdx.x * 256 + threadIdx.x;
    if (i >= n4) return;
    float4 f = ((const float4*)src)[i];
    v4h h; h[0]=(_Float16)f.x; h[1]=(_Float16)f.y; h[2]=(_Float16)f.z; h[3]=(_Float16)f.w;
    ((v4h*)dst)[i] = h;
}

// ---------------------------------------------------------------- kernel 2
// Y = X @ W^T + b  (X fp32 [BS,1024], W f16 [n][k] 1024x1024)
// vmode 0: out[((b*H+h)*S + s)*64 + d]   (Q, K head-major)
// vmode 1: out[((b*H+h)*64 + d)*S + s]   (V transposed: dim-major)
// Block 256 thr (8 waves), block tile 128(M) x 128(N).
// Wave tile 32(M) x 64(N): 2 A-frags x 4 B-frags -> 8 WMMA per k-step, double buffered.
__global__ __launch_bounds__(256) void qkv_proj_kernel(const float* __restrict__ X,
                                                       const _Float16* __restrict__ Wh,
                                                       const float* __restrict__ bias,
                                                       _Float16* __restrict__ out, int vmode) {
    const int l = threadIdx.x & 31, w = threadIdx.x >> 5;
    const int mbase = blockIdx.x * 128 + (w & 3) * 32;
    const int nbase = blockIdx.y * 128 + (w >> 2) * 64;

    v8f acc[2][4];
#pragma unroll
    for (int i = 0; i < 2; ++i)
#pragma unroll
        for (int t = 0; t < 4; ++t) acc[i][t] = (v8f)(0.f);

    v16h a[2], b[4];
#pragma unroll
    for (int i = 0; i < 2; ++i)
        a[i] = frag_a_f32(X + (size_t)(mbase + i * 16) * D_MODEL, D_MODEL, l);
#pragma unroll
    for (int t = 0; t < 4; ++t)
        b[t] = frag_b_f16(Wh + (size_t)(nbase + t * 16) * D_MODEL, D_MODEL, l);

    for (int ks = 0; ks < D_MODEL / 32 - 1; ++ks) {
        const int kkn = (ks + 1) * 32;
        v16h an[2], bn[4];
#pragma unroll
        for (int i = 0; i < 2; ++i)
            an[i] = frag_a_f32(X + (size_t)(mbase + i * 16) * D_MODEL + kkn, D_MODEL, l);
#pragma unroll
        for (int t = 0; t < 4; ++t)
            bn[t] = frag_b_f16(Wh + (size_t)(nbase + t * 16) * D_MODEL + kkn, D_MODEL, l);
#pragma unroll
        for (int i = 0; i < 2; ++i)
#pragma unroll
            for (int t = 0; t < 4; ++t) acc[i][t] = wmma_f16(a[i], b[t], acc[i][t]);
#pragma unroll
        for (int i = 0; i < 2; ++i) a[i] = an[i];
#pragma unroll
        for (int t = 0; t < 4; ++t) b[t] = bn[t];
    }
#pragma unroll
    for (int i = 0; i < 2; ++i)
#pragma unroll
        for (int t = 0; t < 4; ++t) acc[i][t] = wmma_f16(a[i], b[t], acc[i][t]);

    const int gr = l >> 4, cl = l & 15;
#pragma unroll
    for (int t = 0; t < 4; ++t) {
        const int col = nbase + t * 16 + cl;
        const float bv = bias[col];
        const int h = col >> 6, d = col & 63;
#pragma unroll
        for (int i = 0; i < 2; ++i) {
#pragma unroll
            for (int r = 0; r < 8; ++r) {
                const int row = mbase + i * 16 + r + 8 * gr;     // global (b,s) row
                const int bI = row >> 11, s = row & (SEQ - 1);
                const float v = acc[i][t][r] + bv;
                size_t idx;
                if (vmode == 0)
                    idx = (((size_t)(bI * N_HEAD + h) * SEQ) + s) * D_K + d;
                else
                    idx = (((size_t)(bI * N_HEAD + h) * D_K) + d) * SEQ + s;
                out[idx] = (_Float16)v;
            }
        }
    }
}

// ---------------------------------------------------------------- kernel 3
// Flash attention. Grid (B*H, S/64), block 128 (4 waves). Wave owns 16 query rows.
// Qh,Kh: [BH, S, 64] f16 ; Vt: [BH, 64, S] f16 ; AttO: [B, S, 1024] f16
// K-fragments for key-tile kt+1 are prefetched right after the score WMMAs so the
// global loads overlap the softmax exp2/shfl VALU work.
__global__ __launch_bounds__(128) void attn_kernel(const _Float16* __restrict__ Qh,
                                                   const _Float16* __restrict__ Kh,
                                                   const _Float16* __restrict__ Vt,
                                                   _Float16* __restrict__ AttO) {
    const int bh = blockIdx.x;               // b*16 + h
    const int l = threadIdx.x & 31, w = threadIdx.x >> 5;
    const int qbase = blockIdx.y * 64 + w * 16;

    const _Float16* Qp = Qh + (size_t)bh * SEQ * D_K;
    const _Float16* Kp = Kh + (size_t)bh * SEQ * D_K;
    const _Float16* Vp = Vt + (size_t)bh * D_K * SEQ;

    // Q fragments: dims 0..31 and 32..63
    const v16h aq0 = frag_a_f16(Qp + (size_t)qbase * D_K + 0,  D_K, l);
    const v16h aq1 = frag_a_f16(Qp + (size_t)qbase * D_K + 32, D_K, l);

    v8f o[4];
#pragma unroll
    for (int n = 0; n < 4; ++n) o[n] = (v8f)(0.f);
    float mrow[8], lrow[8];
#pragma unroll
    for (int r = 0; r < 8; ++r) { mrow[r] = -3.0e38f; lrow[r] = 0.f; }

    __shared__ __align__(16) _Float16 pbuf[4][16][40];   // per-wave 16x32 P tile (+pad)

    const float c = 0.125f * 1.44269504088896f;          // 1/sqrt(64) * log2(e)
    const int gr = l >> 4, cl = l & 15;

    // K-side B fragments for key-tile 0: {keys 0-15, 16-31} x {dims 0-31, 32-63}
    v16h kb[4];
    kb[0] = frag_b_f16(Kp + 0 * D_K + 0,  D_K, l);
    kb[1] = frag_b_f16(Kp + 0 * D_K + 32, D_K, l);
    kb[2] = frag_b_f16(Kp + 16 * D_K + 0,  D_K, l);
    kb[3] = frag_b_f16(Kp + 16 * D_K + 32, D_K, l);

    for (int kt = 0; kt < SEQ / 32; ++kt) {
        const int kb0 = kt * 32;
        // scores: S0 = keys kb0..+15, S1 = keys kb0+16..+31 ; K-dim = 64 (2 wmma steps)
        v8f s0 = (v8f)(0.f), s1 = (v8f)(0.f);
        s0 = wmma_f16(aq0, kb[0], s0);
        s0 = wmma_f16(aq1, kb[1], s0);
        s1 = wmma_f16(aq0, kb[2], s1);
        s1 = wmma_f16(aq1, kb[3], s1);

        // prefetch next key tile's K fragments (hidden under softmax VALU)
        if (kt < SEQ / 32 - 1) {
            const int nk = kb0 + 32;
            kb[0] = frag_b_f16(Kp + (size_t)(nk + 0) * D_K + 0,  D_K, l);
            kb[1] = frag_b_f16(Kp + (size_t)(nk + 0) * D_K + 32, D_K, l);
            kb[2] = frag_b_f16(Kp + (size_t)(nk + 16) * D_K + 0,  D_K, l);
            kb[3] = frag_b_f16(Kp + (size_t)(nk + 16) * D_K + 32, D_K, l);
        }

        float alpha[8];
#pragma unroll
        for (int r = 0; r < 8; ++r) {
            float a0 = s0[r] * c, a1 = s1[r] * c;
            float mx = fmaxf(a0, a1);
#pragma unroll
            for (int msk = 1; msk <= 8; msk <<= 1) mx = fmaxf(mx, __shfl_xor(mx, msk, 32));
            const float mnew = fmaxf(mrow[r], mx);
            alpha[r] = exp2f(mrow[r] - mnew);
            mrow[r] = mnew;
            a0 = exp2f(a0 - mnew); a1 = exp2f(a1 - mnew);
            s0[r] = a0; s1[r] = a1;
            float ps = a0 + a1;
#pragma unroll
            for (int msk = 1; msk <= 8; msk <<= 1) ps += __shfl_xor(ps, msk, 32);
            lrow[r] = lrow[r] * alpha[r] + ps;
        }
        // rescale running output
#pragma unroll
        for (int n = 0; n < 4; ++n)
#pragma unroll
            for (int r = 0; r < 8; ++r) o[n][r] *= alpha[r];
        // P (C-layout) -> LDS -> A-layout fragment
#pragma unroll
        for (int r = 0; r < 8; ++r) {
            pbuf[w][r + 8 * gr][cl]      = (_Float16)s0[r];
            pbuf[w][r + 8 * gr][16 + cl] = (_Float16)s1[r];
        }
        asm volatile("s_wait_dscnt 0" ::: "memory");
        v16h ap;
        {
            const int row = l & 15, kbp = gr * 8;
            v8h lo = *(const v8h*)&pbuf[w][row][kbp];
            v8h hi = *(const v8h*)&pbuf[w][row][kbp + 16];
#pragma unroll
            for (int i = 0; i < 8; ++i) { ap[i] = lo[i]; ap[i + 8] = hi[i]; }
        }
        // O += P @ V : B fragment from Vt[dim][key], contiguous in key
#pragma unroll
        for (int n = 0; n < 4; ++n) {
            v16h bv = frag_b_f16(Vp + (size_t)(n * 16) * SEQ + kb0, SEQ, l);
            o[n] = wmma_f16(ap, bv, o[n]);
        }
    }

    // normalize and store [B,S,1024] f16
    const int bI = bh >> 4, h = bh & 15;
#pragma unroll
    for (int r = 0; r < 8; ++r) {
        const float inv = 1.0f / lrow[r];
        const int s = qbase + r + 8 * gr;
        const size_t base = ((size_t)bI * SEQ + s) * D_MODEL + h * D_K;
#pragma unroll
        for (int n = 0; n < 4; ++n)
            AttO[base + n * 16 + cl] = (_Float16)(o[n][r] * inv);
    }
}

// ---------------------------------------------------------------- kernel 4
// out = LN( AttO @ Wo^T + bo + q ) * gamma + beta
// Grid BS/16 blocks of 256 thr (8 waves); wave w owns cols [w*128, w*128+128).
// A fragment double buffered; the 8 B fragments are batch-loaded (clause) per k-step.
__global__ __launch_bounds__(256) void oproj_ln_kernel(const _Float16* __restrict__ AttO,
                                                       const _Float16* __restrict__ Who,
                                                       const float* __restrict__ bo,
                                                       const float* __restrict__ resid,
                                                       const float* __restrict__ gamma,
                                                       const float* __restrict__ beta,
                                                       float* __restrict__ out) {
    const int l = threadIdx.x & 31, w = threadIdx.x >> 5;
    const int mbase = blockIdx.x * 16;
    const int cbase = w * 128;
    const int gr = l >> 4, cl = l & 15;

    v8f acc[8];
#pragma unroll
    for (int t = 0; t < 8; ++t) acc[t] = (v8f)(0.f);

    v16h a = frag_a_f16(AttO + (size_t)mbase * D_MODEL, D_MODEL, l);
    for (int ks = 0; ks < D_MODEL / 32; ++ks) {
        const int kk = ks * 32;
        v16h an;
        if (ks < D_MODEL / 32 - 1)
            an = frag_a_f16(AttO + (size_t)mbase * D_MODEL + kk + 32, D_MODEL, l);
        v16h b[8];
#pragma unroll
        for (int t = 0; t < 8; ++t)
            b[t] = frag_b_f16(Who + (size_t)(cbase + t * 16) * D_MODEL + kk, D_MODEL, l);
#pragma unroll
        for (int t = 0; t < 8; ++t) acc[t] = wmma_f16(a, b[t], acc[t]);
        if (ks < D_MODEL / 32 - 1) a = an;
    }

    // bias + residual, then per-row stats
#pragma unroll
    for (int t = 0; t < 8; ++t) {
        const int col = cbase + t * 16 + cl;
        const float bv = bo[col];
#pragma unroll
        for (int r = 0; r < 8; ++r) {
            const int row = mbase + r + 8 * gr;
            acc[t][r] += bv + resid[(size_t)row * D_MODEL + col];
        }
    }

    __shared__ float redbuf[8][16][2];
    __shared__ float mv[16][2];
#pragma unroll
    for (int r = 0; r < 8; ++r) {
        float s = 0.f, q2 = 0.f;
#pragma unroll
        for (int t = 0; t < 8; ++t) { s += acc[t][r]; q2 += acc[t][r] * acc[t][r]; }
#pragma unroll
        for (int msk = 1; msk <= 8; msk <<= 1) {
            s  += __shfl_xor(s,  msk, 32);
            q2 += __shfl_xor(q2, msk, 32);
        }
        if (cl == 0) { redbuf[w][r + 8 * gr][0] = s; redbuf[w][r + 8 * gr][1] = q2; }
    }
    __syncthreads();
    if (threadIdx.x < 16) {
        float s = 0.f, q2 = 0.f;
#pragma unroll
        for (int ww = 0; ww < 8; ++ww) { s += redbuf[ww][threadIdx.x][0]; q2 += redbuf[ww][threadIdx.x][1]; }
        const float mean = s * (1.0f / D_MODEL);
        const float var = q2 * (1.0f / D_MODEL) - mean * mean;
        mv[threadIdx.x][0] = mean;
        mv[threadIdx.x][1] = rsqrtf(var + 1e-5f);
    }
    __syncthreads();
#pragma unroll
    for (int t = 0; t < 8; ++t) {
        const int col = cbase + t * 16 + cl;
        const float g = gamma[col], bt = beta[col];
#pragma unroll
        for (int r = 0; r < 8; ++r) {
            const int rl = r + 8 * gr;
            const float v = (acc[t][r] - mv[rl][0]) * mv[rl][1] * g + bt;
            out[(size_t)(mbase + rl) * D_MODEL + col] = v;
        }
    }
}

// ---------------------------------------------------------------- launcher
extern "C" void kernel_launch(void* const* d_in, const int* in_sizes, int n_in,
                              void* d_out, int out_size, void* d_ws, size_t ws_size,
                              hipStream_t stream) {
    const float* q  = (const float*)d_in[0];
    const float* k  = (const float*)d_in[1];
    const float* v  = (const float*)d_in[2];
    const float* Wq = (const float*)d_in[3];
    const float* bq = (const float*)d_in[4];
    const float* Wk = (const float*)d_in[5];
    const float* bk = (const float*)d_in[6];
    const float* Wv = (const float*)d_in[7];
    const float* bv = (const float*)d_in[8];
    const float* Wo = (const float*)d_in[9];
    const float* bo = (const float*)d_in[10];
    const float* gamma = (const float*)d_in[11];
    const float* beta  = (const float*)d_in[12];
    float* out = (float*)d_out;

    // workspace layout (f16): 4 weight matrices (1M each) + Qh,Kh,Vt,AttO (8M each) = 72 MB
    _Float16* Whq = (_Float16*)d_ws;
    _Float16* Whk = Whq + (size_t)D_MODEL * D_MODEL;
    _Float16* Whv = Whk + (size_t)D_MODEL * D_MODEL;
    _Float16* Who = Whv + (size_t)D_MODEL * D_MODEL;
    _Float16* Qh  = Who + (size_t)D_MODEL * D_MODEL;
    _Float16* Kh  = Qh  + (size_t)BS * D_MODEL;
    _Float16* Vt  = Kh  + (size_t)BS * D_MODEL;
    _Float16* AttO = Vt + (size_t)BS * D_MODEL;

    const int n4 = (D_MODEL * D_MODEL) / 4;
    const int cblk = (n4 + 255) / 256;
    convert_w_kernel<<<cblk, 256, 0, stream>>>(Wq, Whq, n4);
    convert_w_kernel<<<cblk, 256, 0, stream>>>(Wk, Whk, n4);
    convert_w_kernel<<<cblk, 256, 0, stream>>>(Wv, Whv, n4);
    convert_w_kernel<<<cblk, 256, 0, stream>>>(Wo, Who, n4);

    dim3 gproj(BS / 128, D_MODEL / 128);
    qkv_proj_kernel<<<gproj, 256, 0, stream>>>(q, Whq, bq, Qh, 0);
    qkv_proj_kernel<<<gproj, 256, 0, stream>>>(k, Whk, bk, Kh, 0);
    qkv_proj_kernel<<<gproj, 256, 0, stream>>>(v, Whv, bv, Vt, 1);

    dim3 gattn(BATCH * N_HEAD, SEQ / 64);
    attn_kernel<<<gattn, 128, 0, stream>>>(Qh, Kh, Vt, AttO);

    oproj_ln_kernel<<<BS / 16, 256, 0, stream>>>(AttO, Who, bo, q, gamma, beta, out);
}
